// RNN_RNN_36567351558836
// MI455X (gfx1250) — matrix operand order repaired and checked by reading, not compiled, not run on hardware
//
#include <hip/hip_runtime.h>
#include <math.h>
#include <float.h>

// ---------------- CDNA5 WMMA / TDM types ----------------
typedef __attribute__((ext_vector_type(16))) __bf16 v16bf;
typedef __attribute__((ext_vector_type(8)))  float  v8f;
typedef unsigned int v4u __attribute__((ext_vector_type(4)));
typedef int          v8i __attribute__((ext_vector_type(8)));
typedef int          v4i __attribute__((ext_vector_type(4)));

// ---------------- problem constants ----------------
#define Bn    32
#define Pn    100
#define Lw    50
#define Dw    300
#define Hh    256
#define H2    512
#define H3    768
#define Nsent 3200   /* Bn*Pn */
#define KW    576    /* 300+256 padded to 32 */
#define KS    768    /* 512+256 */
#define NG    1024   /* 4 gate groups x 256 */

// ---------------- GEMM tiling ----------------
#define BM 64
#define BN 128
#define BK 32

__device__ __forceinline__ unsigned short f2bf(float f) {
  union { float f; unsigned u; } v; v.f = f;
  unsigned r = v.u + 0x7FFFu + ((v.u >> 16) & 1u);  // round-to-nearest-even
  return (unsigned short)(r >> 16);
}
__device__ __forceinline__ float sigmoidf_(float x) { return 1.0f / (1.0f + __expf(-x)); }

// =====================================================================
// Weight packing: W' is [K][NG] with 4 gate groups, stored pair-interleaved
// bf16: dword index = (k>>1)*NG + n, lo=k even, hi=k odd.
// =====================================================================
__device__ __forceinline__ float wprime_word(const float* wih, const float* whh, int k, int n) {
  if (n < 512) {                    // r,z: gi+gh fused
    if (k < Dw)      return wih[n * Dw + k];
    if (k < Dw + Hh) return whh[n * Hh + (k - Dw)];
    return 0.f;
  } else if (n < 768) {             // in_ (input part of n gate)
    return (k < Dw) ? wih[n * Dw + k] : 0.f;
  } else {                          // hn (hidden part of n gate)
    return (k >= Dw && k < Dw + Hh) ? whh[(n - 256) * Hh + (k - Dw)] : 0.f;
  }
}
__device__ __forceinline__ float wprime_sent(const float* wih, const float* whh, int k, int n) {
  if (n < 512) {
    if (k < H2) return wih[n * H2 + k];
    return whh[n * Hh + (k - H2)];
  } else if (n < 768) {
    return (k < H2) ? wih[n * H2 + k] : 0.f;
  } else {
    return (k >= H2) ? whh[(n - 256) * Hh + (k - H2)] : 0.f;
  }
}

__global__ void pack_word_kernel(const float* wih_f, const float* whh_f,
                                 const float* wih_b, const float* whh_b,
                                 unsigned* out) {
  int idx = blockIdx.x * blockDim.x + threadIdx.x;
  const int per = (KW / 2) * NG;
  if (idx >= 2 * per) return;
  int dir = idx / per, r = idx % per;
  int kp = r / NG, n = r % NG;
  const float* wih = dir ? wih_b : wih_f;
  const float* whh = dir ? whh_b : whh_f;
  unsigned lo = f2bf(wprime_word(wih, whh, 2 * kp, n));
  unsigned hi = f2bf(wprime_word(wih, whh, 2 * kp + 1, n));
  out[idx] = lo | (hi << 16);
}
__global__ void pack_sent_kernel(const float* wih_f, const float* whh_f,
                                 const float* wih_b, const float* whh_b,
                                 unsigned* out) {
  int idx = blockIdx.x * blockDim.x + threadIdx.x;
  const int per = (KS / 2) * NG;
  if (idx >= 2 * per) return;
  int dir = idx / per, r = idx % per;
  int kp = r / NG, n = r % NG;
  const float* wih = dir ? wih_b : wih_f;
  const float* whh = dir ? whh_b : whh_f;
  unsigned lo = f2bf(wprime_sent(wih, whh, 2 * kp, n));
  unsigned hi = f2bf(wprime_sent(wih, whh, 2 * kp + 1, n));
  out[idx] = lo | (hi << 16);
}
// sent_fc:  C = A @ W^T  ->  B[k][n] = W[n*H2+k]
__global__ void pack_sfc_kernel(const float* w, unsigned* out) {
  int idx = blockIdx.x * blockDim.x + threadIdx.x;
  if (idx >= (H2 / 2) * H2) return;
  int kp = idx / H2, n = idx % H2;
  unsigned lo = f2bf(w[n * H2 + 2 * kp]);
  unsigned hi = f2bf(w[n * H2 + 2 * kp + 1]);
  out[idx] = lo | (hi << 16);
}

// =====================================================================
// Tensor Data Mover: DMA one B tile (16 k-pair rows x BN dwords, pair-
// interleaved bf16 weights) into LDS with 1-dword row padding (->stride 129).
// Descriptor per CDNA5 ISA.
// =====================================================================
__device__ __forceinline__ void tdm_load_b_tile(unsigned lds_addr, const unsigned* gsrc,
                                                int Nw, int rows_total) {
  unsigned long long ga = (unsigned long long)(size_t)gsrc;
  v4u g0;
  g0[0] = 1u;                                   // count=1, user mode
  g0[1] = lds_addr;                             // lds_addr (bytes)
  g0[2] = (unsigned)(ga & 0xFFFFFFFFull);       // global_addr[31:0]
  g0[3] = ((unsigned)(ga >> 32) & 0x01FFFFFFu)  // global_addr[56:32]
          | (2u << 30);                         // type=2 ("image")
  v8i g1;
  g1[0] = (int)((2u << 16)                      // data_size = 4B
                | (1u << 20)                    // pad_enable
                | (6u << 22)                    // pad_interval: 128 dwords
                | (0u << 25));                  // pad_amount: 1 dword
  g1[1] = (int)(((unsigned)Nw & 0xFFFFu) << 16);            // tensor_dim0[15:0]
  g1[2] = (int)((((unsigned)Nw >> 16) & 0xFFFFu)            // tensor_dim0[31:16]
                | (((unsigned)rows_total & 0xFFFFu) << 16)); // tensor_dim1[15:0]
  g1[3] = (int)((((unsigned)rows_total >> 16) & 0xFFFFu)
                | (128u << 16));                // tile_dim0 = 128 dwords
  g1[4] = 16;                                   // tile_dim1 = 16 rows, tile_dim2 = 0
  g1[5] = (int)(unsigned)Nw;                    // tensor_dim0_stride[31:0]
  g1[6] = 0;                                    // stride hi / dim1_stride
  g1[7] = 0;
  v4i z4 = {0, 0, 0, 0};
#if defined(__clang_major__) && (__clang_major__ >= 23)
  v8i z8 = {0, 0, 0, 0, 0, 0, 0, 0};
  __builtin_amdgcn_tensor_load_to_lds(g0, g1, z4, z4, z8, 0);
#else
  __builtin_amdgcn_tensor_load_to_lds(g0, g1, z4, z4, 0);
#endif
}

// =====================================================================
// WMMA GEMM: C[dir][M][Nw] = A[dir] @ B'[dir]   (bf16 x bf16 -> f32)
//   mode 0: A row m = [ embed_w[x[m*Lw + t_dir]] (300, f32->bf16) | h_prev (256, bf16) ]
//   mode 1: A row m = [ sentvec[m*Pn + t_dir]    (512, bf16)      | h_prev (256, bf16) ]
//   mode 2: A row m = [ part1[m]                 (p1w, bf16) ]
// B tiles arrive via TDM (async tensor DMA); A is gathered/converted by VALU.
// =====================================================================
__global__ __launch_bounds__(256) void wmma_gemm_kernel(
    const unsigned* __restrict__ Bp, int K, int Nw, int M, int mode,
    const float* __restrict__ embw, const int* __restrict__ xtok,
    int t_f, int t_b,
    const unsigned short* __restrict__ part1, int p1w,
    const unsigned short* __restrict__ hprev,
    float* __restrict__ C,
    int dsB, int dsH, int dsC)
{
  __shared__ unsigned As_u[BM][20];       // 64 rows x 16 k-pair dwords (stride 20 -> 16B aligned)
  __shared__ unsigned Bs_u[16][BN + 1];   // 16 k-pair rows x 128 n + 1 pad (TDM pad_amount)

  const int tid = threadIdx.x;
  const int dir = blockIdx.z;
  const int m0  = blockIdx.x * BM;
  const int n0  = blockIdx.y * BN;
  const int lane = tid & 31;
  const int wv   = tid >> 5;
  const int wm   = wv & 3;      // 4 M sub-tiles
  const int wn   = wv >> 2;     // 2 N groups of 64

  v8f zero8 = {0.f, 0.f, 0.f, 0.f, 0.f, 0.f, 0.f, 0.f};
  v8f acc[4] = {zero8, zero8, zero8, zero8};

  const int lr = tid >> 2;          // A row this thread loads
  const int ks = (tid & 3) * 8;     // A k start within tile

  const int tdir = dir ? t_b : t_f;
  const unsigned lds_b_addr = (unsigned)(size_t)(void*)&Bs_u[0][0];

  // ---- loop-invariant per-thread A source pointers ----
  const int m = m0 + lr;
  const float* erow = nullptr;
  const unsigned short* p1 = nullptr;
  const unsigned short* hrow = nullptr;
  if (m < M) {
    if (mode == 0) {
      const int tok = xtok[m * Lw + tdir];
      erow = embw + (long)tok * Dw;
      __builtin_prefetch(erow, 0, 3);             // global_prefetch_b8
      hrow = hprev + (long)dir * dsH + (long)m * Hh;
    } else if (mode == 1) {
      p1 = part1 + ((long)(m * Pn + tdir)) * H2;
      hrow = hprev + (long)dir * dsH + (long)m * Hh;
    } else {
      p1 = part1 + (long)m * p1w;
    }
  }

  for (int kk = 0; kk < K; kk += BK) {
    // ---- wave 0: kick off async TDM for this B tile ----
    if (wv == 0) {
      const unsigned* gsrc = Bp + (long)dir * dsB + (long)(kk >> 1) * Nw + n0;
      tdm_load_b_tile(lds_b_addr, gsrc, Nw, K >> 1);
    }

    // ---- all threads: stage A tile into LDS (gather + f32->bf16) ----
    unsigned short tv[8];
    if (m < M) {
      if (mode == 0) {
        #pragma unroll
        for (int i = 0; i < 8; i++) {
          int k = kk + ks + i;
          tv[i] = (k < Dw)      ? f2bf(erow[k])
                : (k < Dw + Hh) ? hrow[k - Dw]
                : (unsigned short)0;
        }
      } else if (mode == 1) {
        #pragma unroll
        for (int i = 0; i < 8; i++) {
          int k = kk + ks + i;
          tv[i] = (k < H2) ? p1[k] : hrow[k - H2];
        }
      } else {
        #pragma unroll
        for (int i = 0; i < 8; i++) {
          int k = kk + ks + i;
          tv[i] = (k < p1w) ? p1[k] : (unsigned short)0;
        }
      }
    } else {
      #pragma unroll
      for (int i = 0; i < 8; i++) tv[i] = 0;
    }
    {
      uint4 pk;
      pk.x = (unsigned)tv[0] | ((unsigned)tv[1] << 16);
      pk.y = (unsigned)tv[2] | ((unsigned)tv[3] << 16);
      pk.z = (unsigned)tv[4] | ((unsigned)tv[5] << 16);
      pk.w = (unsigned)tv[6] | ((unsigned)tv[7] << 16);
      *(uint4*)&As_u[lr][ks >> 1] = pk;         // ds_store_b128 (16B aligned)
    }

    // ---- wave 0 drains its tensor counter, then block-wide release ----
    if (wv == 0) __builtin_amdgcn_s_wait_tensorcnt(0);
    __syncthreads();

    // ---- build fragments + WMMA ----
    union { v16bf v; unsigned u[8]; } afr;
    const int fm = wm * 16 + (lane & 15);
    const int kd = (lane < 16) ? 0 : 8;   // A k-pair offset per lane half
    {
      uint4 lo = *(const uint4*)&As_u[fm][kd];       // ds_load_b128
      uint4 hi = *(const uint4*)&As_u[fm][kd + 4];   // ds_load_b128
      afr.u[0] = lo.x; afr.u[1] = lo.y; afr.u[2] = lo.z; afr.u[3] = lo.w;
      afr.u[4] = hi.x; afr.u[5] = hi.y; afr.u[6] = hi.z; afr.u[7] = hi.w;
    }
    const int kr = (lane < 16) ? 0 : 8;   // B k-pair row base per lane half
    #pragma unroll
    for (int s = 0; s < 4; s++) {
      union { v16bf v; unsigned u[8]; } bfr;
      const int bnn = wn * 64 + s * 16 + (lane & 15);
      #pragma unroll
      for (int v2 = 0; v2 < 8; v2++) bfr.u[v2] = Bs_u[kr + v2][bnn];
      acc[s] = __builtin_amdgcn_wmma_f32_16x16x32_bf16(
                 false, afr.v, false, bfr.v, (short)0, acc[s], false, false);
    }
    __syncthreads();
  }

  // ---- epilogue: C fragment layout (lane&15 = n, VGPR i -> m, +8 for hi lanes) ----
  const int rbase = m0 + wm * 16 + ((lane < 16) ? 0 : 8);
  const int cn = lane & 15;
  #pragma unroll
  for (int s = 0; s < 4; s++) {
    const int col = n0 + wn * 64 + s * 16 + cn;
    #pragma unroll
    for (int i = 0; i < 8; i++) {
      const int row = rbase + i;
      if (row < M) C[(long)dir * dsC + (long)row * Nw + col] = acc[s][i];
    }
  }
}

// =====================================================================
// GRU gate math + hidden update + running masked max-pool
// G row layout: [ir+hr(256) | iz+hz(256) | in(256) | hn(256)]
// =====================================================================
__global__ void word_gates_kernel(const float* __restrict__ G,
                                  const float* bih_f, const float* bhh_f,
                                  const float* bih_b, const float* bhh_b,
                                  float* __restrict__ h32,
                                  unsigned short* __restrict__ hbf,
                                  float* __restrict__ pooled,
                                  const int* __restrict__ slen, int t)
{
  int idx = blockIdx.x * blockDim.x + threadIdx.x;
  if (idx >= 2 * Nsent * Hh) return;
  int dir = idx / (Nsent * Hh);
  int r0 = idx % (Nsent * Hh);
  int m = r0 / Hh, j = r0 % Hh;
  const float* g = G + ((long)dir * Nsent + m) * NG;
  const float* bih = dir ? bih_b : bih_f;
  const float* bhh = dir ? bhh_b : bhh_f;
  float rg = sigmoidf_(g[j] + bih[j] + bhh[j]);
  float zg = sigmoidf_(g[Hh + j] + bih[Hh + j] + bhh[Hh + j]);
  float nn = tanhf(g[2 * Hh + j] + bih[2 * Hh + j] + rg * (g[3 * Hh + j] + bhh[2 * Hh + j]));
  long hidx = (long)dir * Nsent * Hh + (long)m * Hh + j;
  float h = (1.f - zg) * nn + zg * h32[hidx];
  h32[hidx] = h;
  hbf[hidx] = f2bf(h);
  int tdir = dir ? (Lw - 1 - t) : t;
  if (tdir < slen[m]) {
    long pidx = (long)m * H2 + dir * Hh + j;
    pooled[pidx] = fmaxf(pooled[pidx], h);
  }
}

__global__ void sent_gates_kernel(const float* __restrict__ G,
                                  const float* bih_f, const float* bhh_f,
                                  const float* bih_b, const float* bhh_b,
                                  float* __restrict__ h32,
                                  unsigned short* __restrict__ hbf,
                                  float* __restrict__ shmax)
{
  int idx = blockIdx.x * blockDim.x + threadIdx.x;
  if (idx >= 2 * Bn * Hh) return;
  int dir = idx / (Bn * Hh);
  int r0 = idx % (Bn * Hh);
  int m = r0 / Hh, j = r0 % Hh;
  const float* g = G + ((long)dir * Bn + m) * NG;
  const float* bih = dir ? bih_b : bih_f;
  const float* bhh = dir ? bhh_b : bhh_f;
  float rg = sigmoidf_(g[j] + bih[j] + bhh[j]);
  float zg = sigmoidf_(g[Hh + j] + bih[Hh + j] + bhh[Hh + j]);
  float nn = tanhf(g[2 * Hh + j] + bih[2 * Hh + j] + rg * (g[3 * Hh + j] + bhh[2 * Hh + j]));
  long hidx = (long)dir * Bn * Hh + (long)m * Hh + j;
  float h = (1.f - zg) * nn + zg * h32[hidx];
  h32[hidx] = h;
  hbf[hidx] = f2bf(h);
  long pidx = (long)m * H2 + dir * Hh + j;
  shmax[pidx] = fmaxf(shmax[pidx], h);     // full-length pool
}

// ---------------- small helpers ----------------
__global__ void slen_kernel(const int* __restrict__ x, int* __restrict__ slen) {
  int m = blockIdx.x * blockDim.x + threadIdx.x;
  if (m >= Nsent) return;
  int c = 0;
  for (int i = 0; i < Lw; i++) c += (x[m * Lw + i] > 0);
  slen[m] = c;
}
__global__ void sentvec_kernel(const float* __restrict__ pooled, const int* __restrict__ slen,
                               unsigned short* __restrict__ svbf) {
  int idx = blockIdx.x * blockDim.x + threadIdx.x;
  if (idx >= Nsent * H2) return;
  int m = idx / H2;
  float v = (slen[m] > 0) ? pooled[idx] : 0.f;
  svbf[idx] = f2bf(v);
}
__global__ void tanh_bias_kernel(const float* __restrict__ g, const float* __restrict__ b,
                                 float* __restrict__ out, int n) {
  int i = blockIdx.x * blockDim.x + threadIdx.x;
  if (i < n) out[i] = tanhf(g[i] + b[i % H2]);
}
__global__ void matvec_kernel(const float* __restrict__ in, const float* __restrict__ W,
                              const float* __restrict__ bias, float* __restrict__ out,
                              int do_tanh) {
  int idx = blockIdx.x * blockDim.x + threadIdx.x;
  if (idx >= Bn * H2) return;
  int bb = idx / H2, d = idx % H2;
  float a = bias ? bias[d] : 0.f;
  const float* wr = W + (long)d * H2;
  const float* v = in + (long)bb * H2;
  for (int e = 0; e < H2; e++) a += wr[e] * v[e];
  out[idx] = do_tanh ? tanhf(a) : a;
}
__global__ void base_kernel(const float* __restrict__ sent_t, const float* __restrict__ w_content,
                            const float* __restrict__ csal,
                            const float* __restrict__ abs_emb, const float* __restrict__ w_abs,
                            const float* __restrict__ rel_emb, const float* __restrict__ w_rel,
                            const float* __restrict__ sent_bias, const int* __restrict__ blog_len,
                            float* __restrict__ base) {
  int m = blockIdx.x * blockDim.x + threadIdx.x;
  if (m >= Nsent) return;
  int b = m / Pn, p = m % Pn;
  const float* st = sent_t + (long)m * H2;
  const float* cs = csal + (long)b * H2;
  float s1 = 0.f, s2 = 0.f;
  for (int d = 0; d < H2; d++) { float v = st[d]; s1 += v * w_content[d]; s2 += v * cs[d]; }
  float at = 0.f;
  const float* ae = abs_emb + p * 50;
  for (int i = 0; i < 50; i++) at += ae[i] * w_abs[i];
  int ridx = (p * 10) / blog_len[0];
  if (ridx > 9) ridx = 9;
  float rt = 0.f;
  const float* re = rel_emb + ridx * 50;
  for (int i = 0; i < 50; i++) rt += re[i] * w_rel[i];
  base[m] = s1 + s2 + at + rt + sent_bias[0];
}

// Sequential MMR scan: one workgroup per blog, 100 steps, all state in LDS.
__global__ __launch_bounds__(256) void scan_kernel(const float* __restrict__ sent_t,
                                                   const float* __restrict__ Wnov,
                                                   const float* __restrict__ base,
                                                   float* __restrict__ out) {
  __shared__ float s[H2];
  __shared__ float sp[H2];
  __shared__ float red[256];
  __shared__ float prob_sh;
  const int b = blockIdx.x, tid = threadIdx.x;
  for (int c = tid; c < H2; c += 256) s[c] = 0.f;
  __syncthreads();
  for (int p = 0; p < Pn; p++) {
    const float* row = sent_t + ((long)(b * Pn + p)) * H2;
    for (int c = tid; c < H2; c += 256) sp[c] = row[c];
    __syncthreads();
    float a = 0.f;
    for (int d = tid; d < H2; d += 256) {
      const float* wr = Wnov + (long)d * H2;
      float dd = 0.f;
      for (int e = 0; e < H2; e++) dd += wr[e] * s[e];
      a += sp[d] * dd;
    }
    red[tid] = a;
    __syncthreads();
    for (int off = 128; off > 0; off >>= 1) {
      if (tid < off) red[tid] += red[tid + off];
      __syncthreads();
    }
    if (tid == 0) prob_sh = sigmoidf_(base[b * Pn + p] - red[0]);  // nov = -s_p^T Wnov s
    __syncthreads();
    const float pr = prob_sh;
    for (int c = tid; c < H2; c += 256) s[c] += pr * sp[c];
    if (tid == 0) out[b * Pn + p] = pr;
    __syncthreads();
  }
}

// ---------------- fills ----------------
__global__ void fill_f32_kernel(float* p, float v, int n) {
  int i = blockIdx.x * blockDim.x + threadIdx.x;
  if (i < n) p[i] = v;
}
__global__ void fill_u16_kernel(unsigned short* p, unsigned short v, int n) {
  int i = blockIdx.x * blockDim.x + threadIdx.x;
  if (i < n) p[i] = v;
}

// =====================================================================
extern "C" void kernel_launch(void* const* d_in, const int* in_sizes, int n_in,
                              void* d_out, int out_size, void* d_ws, size_t ws_size,
                              hipStream_t stream) {
  (void)in_sizes; (void)n_in; (void)out_size; (void)ws_size;

  const int*   x          = (const int*)  d_in[0];
  const int*   blog_len   = (const int*)  d_in[1];
  const float* embed_w    = (const float*)d_in[2];
  const float* abs_emb    = (const float*)d_in[3];
  const float* rel_emb    = (const float*)d_in[4];
  const float* wgru_wih_f = (const float*)d_in[5];
  const float* wgru_whh_f = (const float*)d_in[6];
  const float* wgru_bih_f = (const float*)d_in[7];
  const float* wgru_bhh_f = (const float*)d_in[8];
  const float* wgru_wih_b = (const float*)d_in[9];
  const float* wgru_whh_b = (const float*)d_in[10];
  const float* wgru_bih_b = (const float*)d_in[11];
  const float* wgru_bhh_b = (const float*)d_in[12];
  const float* sgru_wih_f = (const float*)d_in[13];
  const float* sgru_whh_f = (const float*)d_in[14];
  const float* sgru_bih_f = (const float*)d_in[15];
  const float* sgru_bhh_f = (const float*)d_in[16];
  const float* sgru_wih_b = (const float*)d_in[17];
  const float* sgru_whh_b = (const float*)d_in[18];
  const float* sgru_bih_b = (const float*)d_in[19];
  const float* sgru_bhh_b = (const float*)d_in[20];
  const float* w_content  = (const float*)d_in[21];
  const float* W_sal      = (const float*)d_in[22];
  const float* W_nov      = (const float*)d_in[23];
  const float* w_abs      = (const float*)d_in[24];
  const float* w_rel      = (const float*)d_in[25];
  const float* sent_bias  = (const float*)d_in[26];
  const float* blog_fc_w  = (const float*)d_in[27];
  const float* blog_fc_b  = (const float*)d_in[28];
  const float* sent_fc_w  = (const float*)d_in[29];
  const float* sent_fc_b  = (const float*)d_in[30];

  // ---- workspace carve-out ----
  size_t off = 0;
  char* wsb = (char*)d_ws;
  auto alloc = [&](size_t bytes) -> void* {
    void* p = wsb + off;
    off = (off + bytes + 255) & ~(size_t)255;
    return p;
  };
  unsigned*       Wword  = (unsigned*)alloc(2ull * (KW / 2) * NG * 4);
  unsigned*       Wsent  = (unsigned*)alloc(2ull * (KS / 2) * NG * 4);
  unsigned*       Wsfc   = (unsigned*)alloc((size_t)(H2 / 2) * H2 * 4);
  int*            slen   = (int*)alloc((size_t)Nsent * 4);
  float*          hw32   = (float*)alloc(2ull * Nsent * Hh * 4);
  unsigned short* hwbf   = (unsigned short*)alloc(2ull * Nsent * Hh * 2);
  float*          pooled = (float*)alloc((size_t)Nsent * H2 * 4);
  float*          Gw     = (float*)alloc(2ull * Nsent * NG * 4);
  unsigned short* svbf   = (unsigned short*)alloc((size_t)Nsent * H2 * 2);
  float*          hs32   = (float*)alloc(2ull * Bn * Hh * 4);
  unsigned short* hsbf   = (unsigned short*)alloc(2ull * Bn * Hh * 2);
  float*          Gs     = (float*)alloc(2ull * Bn * NG * 4);
  float*          shmax  = (float*)alloc((size_t)Bn * H2 * 4);
  float*          ctx    = (float*)alloc((size_t)Bn * H2 * 4);
  float*          csal   = (float*)alloc((size_t)Bn * H2 * 4);
  float*          Gfc    = (float*)alloc((size_t)Nsent * H2 * 4);
  float*          sentt  = (float*)alloc((size_t)Nsent * H2 * 4);
  float*          baseb  = (float*)alloc((size_t)Nsent * 4);

  const int TB = 256;
  auto blk = [](int n) { return (n + 255) / 256; };

  // ---- pack weights (bf16 pair-interleaved) ----
  pack_word_kernel<<<blk(2 * (KW / 2) * NG), TB, 0, stream>>>(
      wgru_wih_f, wgru_whh_f, wgru_wih_b, wgru_whh_b, Wword);
  pack_sent_kernel<<<blk(2 * (KS / 2) * NG), TB, 0, stream>>>(
      sgru_wih_f, sgru_whh_f, sgru_wih_b, sgru_whh_b, Wsent);
  pack_sfc_kernel<<<blk((H2 / 2) * H2), TB, 0, stream>>>(sent_fc_w, Wsfc);
  slen_kernel<<<blk(Nsent), TB, 0, stream>>>(x, slen);

  // ---- init word-level state ----
  fill_f32_kernel<<<blk(2 * Nsent * Hh), TB, 0, stream>>>(hw32, 0.f, 2 * Nsent * Hh);
  fill_u16_kernel<<<blk(2 * Nsent * Hh), TB, 0, stream>>>(hwbf, 0, 2 * Nsent * Hh);
  fill_f32_kernel<<<blk(Nsent * H2), TB, 0, stream>>>(pooled, -FLT_MAX, Nsent * H2);

  // ---- word BiGRU: 50 steps, both directions per launch ----
  dim3 gw(Nsent / BM, NG / BN, 2);
  for (int t = 0; t < Lw; t++) {
    wmma_gemm_kernel<<<gw, TB, 0, stream>>>(
        Wword, KW, NG, Nsent, /*mode=*/0,
        embed_w, x, t, Lw - 1 - t,
        nullptr, 0, hwbf, Gw,
        (KW / 2) * NG, Nsent * Hh, Nsent * NG);
    word_gates_kernel<<<blk(2 * Nsent * Hh), TB, 0, stream>>>(
        Gw, wgru_bih_f, wgru_bhh_f, wgru_bih_b, wgru_bhh_b,
        hw32, hwbf, pooled, slen, t);
  }
  sentvec_kernel<<<blk(Nsent * H2), TB, 0, stream>>>(pooled, slen, svbf);

  // ---- init sentence-level state ----
  fill_f32_kernel<<<blk(2 * Bn * Hh), TB, 0, stream>>>(hs32, 0.f, 2 * Bn * Hh);
  fill_u16_kernel<<<blk(2 * Bn * Hh), TB, 0, stream>>>(hsbf, 0, 2 * Bn * Hh);
  fill_f32_kernel<<<blk(Bn * H2), TB, 0, stream>>>(shmax, -FLT_MAX, Bn * H2);

  // ---- sentence BiGRU: 100 steps ----
  dim3 gs(1, NG / BN, 2);
  for (int t = 0; t < Pn; t++) {
    wmma_gemm_kernel<<<gs, TB, 0, stream>>>(
        Wsent, KS, NG, Bn, /*mode=*/1,
        nullptr, nullptr, t, Pn - 1 - t,
        svbf, H2, hsbf, Gs,
        (KS / 2) * NG, Bn * Hh, Bn * NG);
    sent_gates_kernel<<<blk(2 * Bn * Hh), TB, 0, stream>>>(
        Gs, sgru_bih_f, sgru_bhh_f, sgru_bih_b, sgru_bhh_b, hs32, hsbf, shmax);
  }

  // ---- context = tanh(blog_vec @ blog_fc_w^T + b); csal = W_sal @ context ----
  matvec_kernel<<<blk(Bn * H2), TB, 0, stream>>>(shmax, blog_fc_w, blog_fc_b, ctx, 1);
  matvec_kernel<<<blk(Bn * H2), TB, 0, stream>>>(ctx, W_sal, nullptr, csal, 0);

  // ---- sent_t = tanh(sent_vec @ sent_fc_w^T + b)  (WMMA GEMM + epilogue) ----
  dim3 gf(Nsent / BM, H2 / BN, 1);
  wmma_gemm_kernel<<<gf, TB, 0, stream>>>(
      Wsfc, H2, H2, Nsent, /*mode=*/2,
      nullptr, nullptr, 0, 0,
      svbf, H2, nullptr, Gfc, 0, 0, 0);
  tanh_bias_kernel<<<blk(Nsent * H2), TB, 0, stream>>>(Gfc, sent_fc_b, sentt, Nsent * H2);

  // ---- base scores + sequential novelty scan ----
  base_kernel<<<blk(Nsent), TB, 0, stream>>>(
      sentt, w_content, csal, abs_emb, w_abs, rel_emb, w_rel, sent_bias, blog_len, baseb);
  scan_kernel<<<Bn, TB, 0, stream>>>(sentt, W_nov, baseb, (float*)d_out);
}
